// BasicBahdanauAttnDecoder_40183714022189
// MI455X (gfx1250) — compile-verified, matrix operand order checked by reading, not run on hardware
//
#include <hip/hip_runtime.h>
#include <math.h>

// ---------------- dims ----------------
#define TT 64
#define BB 32
#define SS 64
#define VV 32000
#define EE 512
#define HH 512
// x0 width for layer0 = E + 2H = 1536

typedef __attribute__((ext_vector_type(16))) __bf16 v16bf;
typedef __attribute__((ext_vector_type(8)))  float  v8f;

// ---------------- WMMA helpers (bf16, 16x16x32, f32 acc) ----------------
// 16-bit A/B fragment layout (CDNA5 ISA 7.12.2):
//   lane&15  -> row (A) / col (B); lane>>4 -> K half select
//   regs 0..7  hold K = half*8 + 0..7
//   regs 8..15 hold K = 16 + half*8 + 0..7
__device__ __forceinline__ v16bf load_frag16(const __bf16* __restrict__ P, int ld,
                                             int r0, int k0) {
  int lane = threadIdx.x & 31;
  int half = lane >> 4;
  int rr   = lane & 15;
  const __bf16* p = P + (size_t)(r0 + rr) * (size_t)ld + k0 + half * 8;
  v16bf f;
#pragma unroll
  for (int i = 0; i < 8; ++i) f[i] = p[i];
#pragma unroll
  for (int i = 0; i < 8; ++i) f[8 + i] = p[16 + i];
  return f;
}

__device__ __forceinline__ v8f wmma_bf16(v8f c, v16bf a, v16bf b) {
  return __builtin_amdgcn_wmma_f32_16x16x32_bf16(false, a, false, b,
                                                 (short)0, c, false, false);
}

// 4-tile accumulate: acc[t] += A[m0.., :K] * W[n0+t*nstride.., :K]^T
// One A-fragment load feeds 4 WMMAs (register reuse); prefetch 4 panels ahead.
__device__ __forceinline__ void gemm_tiles4(v8f acc[4],
                                            const __bf16* __restrict__ A, int lda,
                                            const __bf16* __restrict__ W, int ldw,
                                            int m0, int n0, int nstride, int K) {
  int lane = threadIdx.x & 31;
  int half = lane >> 4, rr = lane & 15;
  const __bf16* pa = A + (size_t)(m0 + rr) * (size_t)lda + half * 8;
  const __bf16* pw = W + (size_t)(n0 + rr) * (size_t)ldw + half * 8;
  for (int k0 = 0; k0 < K; k0 += 32) {
    __builtin_prefetch(pa + k0 + 128, 0, 1);   // global_prefetch_b8
    __builtin_prefetch(pw + k0 + 128, 0, 1);
    v16bf a = load_frag16(A, lda, m0, k0);
#pragma unroll
    for (int t = 0; t < 4; ++t) {
      v16bf b = load_frag16(W, ldw, n0 + t * nstride, k0);
      acc[t] = wmma_bf16(acc[t], a, b);
    }
  }
}

// D/C tile layout: lane&15 = n, (lane>>4)*8 + r = m
__device__ __forceinline__ void tile_coords(int& n, int& mb) {
  int lane = threadIdx.x & 31;
  n  = lane & 15;
  mb = (lane >> 4) * 8;
}

// ---------------- generic GEMM: O = A @ W^T + bias ----------------
// one wave per 16x64 output strip; grid = (M/16, N/64)
__global__ void gemm_bf16_nt(const __bf16* __restrict__ A,
                             const __bf16* __restrict__ W,
                             const float* __restrict__ bias,
                             float* __restrict__ O,
                             __bf16* __restrict__ Obf,
                             int N, int K) {
  int m0 = blockIdx.x * 16;
  int n0 = blockIdx.y * 64;
  v8f acc[4];
#pragma unroll
  for (int t = 0; t < 4; ++t) acc[t] = (v8f){};
  gemm_tiles4(acc, A, K, W, K, m0, n0, 16, K);
  int n, mb; tile_coords(n, mb);
#pragma unroll
  for (int t = 0; t < 4; ++t) {
    int nn = n0 + t * 16 + n;
    float bv = bias ? bias[nn] : 0.0f;
#pragma unroll
    for (int r = 0; r < 8; ++r) {
      float v = acc[t][r] + bv;
      size_t off = (size_t)(m0 + mb + r) * (size_t)N + nn;
      O[off] = v;
      if (Obf) Obf[off] = (__bf16)v;
    }
  }
}

// ---------------- fused LSTM step ----------------
// gates = [x0|x1] @ Wih^T + h @ Whh^T + bih + bhh ; then cell elementwise.
// grid = (B/16, H/16); one wave owns a 16(batch) x 16(hidden) tile; the four
// gate tiles (i,f,g,o) share each A-fragment load and stay in registers for
// the sigmoid/tanh elementwise.
__global__ void lstm_step(const __bf16* __restrict__ x0, int ldx0, int K0,
                          const __bf16* __restrict__ W0, // Wih cols [0,K0)
                          const __bf16* __restrict__ x1, int ldx1, int K1,
                          const __bf16* __restrict__ W1, // Wih cols [K0,K0+K1)
                          int ldw_ih,
                          const __bf16* __restrict__ hprev,
                          const __bf16* __restrict__ Whh,
                          const float* __restrict__ bih,
                          const float* __restrict__ bhh,
                          const float* __restrict__ c_prev,
                          float* __restrict__ h_out,
                          __bf16* __restrict__ h_out_bf,
                          float* __restrict__ c_out) {
  int m0 = blockIdx.x * 16;
  int j0 = blockIdx.y * 16;
  v8f acc[4];
#pragma unroll
  for (int g = 0; g < 4; ++g) acc[g] = (v8f){};
  gemm_tiles4(acc, x0, ldx0, W0, ldw_ih, m0, j0, HH, K0);
  if (x1) gemm_tiles4(acc, x1, ldx1, W1, ldw_ih, m0, j0, HH, K1);
  gemm_tiles4(acc, hprev, HH, Whh, HH, m0, j0, HH, HH);

  int n, mb; tile_coords(n, mb);
  int j = j0 + n;
  float b_i = bih[j]          + bhh[j];
  float b_f = bih[HH + j]     + bhh[HH + j];
  float b_g = bih[2 * HH + j] + bhh[2 * HH + j];
  float b_o = bih[3 * HH + j] + bhh[3 * HH + j];
#pragma unroll
  for (int r = 0; r < 8; ++r) {
    int m = m0 + mb + r;
    size_t off = (size_t)m * HH + j;
    float gi = acc[0][r] + b_i;
    float gf = acc[1][r] + b_f;
    float gg = acc[2][r] + b_g;
    float go = acc[3][r] + b_o;
    float si = 1.0f / (1.0f + expf(-gi));
    float sf = 1.0f / (1.0f + expf(-gf));
    float so = 1.0f / (1.0f + expf(-go));
    float cn = sf * c_prev[off] + si * tanhf(gg);
    float hn = so * tanhf(cn);
    c_out[off]    = cn;
    h_out[off]    = hn;
    h_out_bf[off] = (__bf16)hn;
  }
}

// ---------------- attention + context + stacked output ----------------
// one block (256 thr) per batch element b
__global__ void attn_step(const float* __restrict__ h1,     // [B,H] new
                          const float* __restrict__ Wh,     // [H,H]
                          const float* __restrict__ bh,
                          const float* __restrict__ annot,  // [S,B,H]
                          const float* __restrict__ Wo,     // [H]
                          const float* __restrict__ bo,
                          const float* __restrict__ enc,    // [S,B,2H]
                          __bf16* __restrict__ ctx_out_bf,  // [B,2H]
                          float* __restrict__ stacked,      // [B,3H] at step t
                          __bf16* __restrict__ stacked_bf) {
  int b = blockIdx.x;
  int tid = threadIdx.x;
  __shared__ float hs[HH];
  __shared__ float ev[SS];

  // hs = h1[b] @ Wh^T + bh
  const float* hrow = h1 + (size_t)b * HH;
  for (int jj = tid; jj < HH; jj += 256) {
    const float* wrow = Wh + (size_t)jj * HH;
    float acc = bh[jj];
    for (int k = 0; k < HH; ++k) acc += hrow[k] * wrow[k];
    hs[jj] = acc;
  }
  __syncthreads();

  // e[s] = Wo . tanh(hs + annot[s,b,:]) + bo
  int wid = tid >> 5, lane = tid & 31;
  for (int s = wid; s < SS; s += 8) {
    const float* arow = annot + ((size_t)s * BB + b) * HH;
    float p = 0.0f;
    for (int jj = lane; jj < HH; jj += 32) p += tanhf(hs[jj] + arow[jj]) * Wo[jj];
#pragma unroll
    for (int off = 16; off > 0; off >>= 1) p += __shfl_down(p, off, 32);
    if (lane == 0) ev[s] = p + bo[0];
  }
  __syncthreads();

  // softmax over S (tiny: one thread)
  if (tid == 0) {
    float mx = ev[0];
    for (int s = 1; s < SS; ++s) mx = fmaxf(mx, ev[s]);
    float sum = 0.0f;
    for (int s = 0; s < SS; ++s) { float x = expf(ev[s] - mx); ev[s] = x; sum += x; }
    float inv = 1.0f / sum;
    for (int s = 0; s < SS; ++s) ev[s] *= inv;
  }
  __syncthreads();

  // ctx[d] = sum_s a[s] * enc[s,b,d]; write stacked = [h1 | ctx]
  for (int d = tid; d < 2 * HH; d += 256) {
    float acc = 0.0f;
    for (int s = 0; s < SS; ++s) acc += ev[s] * enc[((size_t)s * BB + b) * (2 * HH) + d];
    ctx_out_bf[(size_t)b * 2 * HH + d] = (__bf16)acc;
    size_t so = (size_t)b * 3 * HH + HH + d;
    stacked[so]    = acc;
    stacked_bf[so] = (__bf16)acc;
  }
  for (int jj = tid; jj < HH; jj += 256) {
    float v = hrow[jj];
    size_t so = (size_t)b * 3 * HH + jj;
    stacked[so]    = v;
    stacked_bf[so] = (__bf16)v;
  }
}

// ---------------- small utility kernels ----------------
__global__ void f32_to_bf16(const float* __restrict__ s, __bf16* __restrict__ d, size_t n) {
  size_t i = (size_t)blockIdx.x * 256 + threadIdx.x;
  if (i < n) d[i] = (__bf16)s[i];
}

__global__ void gather_emb(const int* __restrict__ trg, const float* __restrict__ emb,
                           __bf16* __restrict__ out) { // [T*B, E]
  int tb = blockIdx.x;
  int idx = trg[tb];
  const float* src = emb + (size_t)idx * EE;
  __bf16* dst = out + (size_t)tb * EE;
  for (int e = threadIdx.x; e < EE; e += 256) dst[e] = (__bf16)src[e];
}

__global__ void copy_f32(const float* __restrict__ s, float* __restrict__ d, size_t n) {
  size_t i = (size_t)blockIdx.x * 256 + threadIdx.x;
  if (i < n) d[i] = s[i];
}

// in-place log_softmax over rows of length N; grid = rows, block = 256
__global__ void log_softmax_rows(float* __restrict__ X, int N) {
  float* x = X + (size_t)blockIdx.x * (size_t)N;
  __shared__ float sred[256];
  int tid = threadIdx.x;
  float mx = -INFINITY;
  for (int i = tid; i < N; i += 256) mx = fmaxf(mx, x[i]);
  sred[tid] = mx; __syncthreads();
  for (int s = 128; s > 0; s >>= 1) {
    if (tid < s) sred[tid] = fmaxf(sred[tid], sred[tid + s]);
    __syncthreads();
  }
  mx = sred[0]; __syncthreads();
  float sum = 0.0f;
  for (int i = tid; i < N; i += 256) sum += expf(x[i] - mx);
  sred[tid] = sum; __syncthreads();
  for (int s = 128; s > 0; s >>= 1) {
    if (tid < s) sred[tid] += sred[tid + s];
    __syncthreads();
  }
  float lse = mx + logf(sred[0]);
  __syncthreads();
  for (int i = tid; i < N; i += 256) x[i] -= lse;
}

// ---------------- host orchestration ----------------
extern "C" void kernel_launch(void* const* d_in, const int* in_sizes, int n_in,
                              void* d_out, int out_size, void* d_ws, size_t ws_size,
                              hipStream_t stream) {
  (void)in_sizes; (void)n_in; (void)out_size; (void)ws_size;

  const int*   trg   = (const int*)  d_in[0];
  const float* enc   = (const float*)d_in[1];
  const float* emb   = (const float*)d_in[2];
  const float* Wih0  = (const float*)d_in[3];
  const float* Whh0  = (const float*)d_in[4];
  const float* bih0  = (const float*)d_in[5];
  const float* bhh0  = (const float*)d_in[6];
  const float* Wih1  = (const float*)d_in[7];
  const float* Whh1  = (const float*)d_in[8];
  const float* bih1  = (const float*)d_in[9];
  const float* bhh1  = (const float*)d_in[10];
  const float* Wa    = (const float*)d_in[11];
  const float* ba    = (const float*)d_in[12];
  const float* Wh    = (const float*)d_in[13];
  const float* bh    = (const float*)d_in[14];
  const float* Wo    = (const float*)d_in[15];
  const float* bo    = (const float*)d_in[16];
  const float* Wp    = (const float*)d_in[17];
  const float* bp    = (const float*)d_in[18];
  const float* b_out = (const float*)d_in[19];

  // ---- carve workspace ----
  char* w = (char*)d_ws;
  auto alloc = [&](size_t bytes) -> void* {
    void* p = (void*)w;
    w += (bytes + 255) & ~(size_t)255;
    return p;
  };
  const int X0K = EE + 2 * HH; // 1536

  __bf16* emb_bf    = (__bf16*)alloc((size_t)VV * EE * 2);
  __bf16* Wih0_bf   = (__bf16*)alloc((size_t)4 * HH * X0K * 2);
  __bf16* Whh0_bf   = (__bf16*)alloc((size_t)4 * HH * HH * 2);
  __bf16* Wih1_bf   = (__bf16*)alloc((size_t)4 * HH * HH * 2);
  __bf16* Whh1_bf   = (__bf16*)alloc((size_t)4 * HH * HH * 2);
  __bf16* Wa_bf     = (__bf16*)alloc((size_t)HH * 2 * HH * 2);
  __bf16* Wp_bf     = (__bf16*)alloc((size_t)EE * 3 * HH * 2);
  __bf16* enc_bf    = (__bf16*)alloc((size_t)SS * BB * 2 * HH * 2);
  __bf16* trgemb_bf = (__bf16*)alloc((size_t)TT * BB * EE * 2);
  float*  annot     = (float*) alloc((size_t)SS * BB * HH * 4);
  float*  stacked   = (float*) alloc((size_t)TT * BB * 3 * HH * 4);
  __bf16* stackedbf = (__bf16*)alloc((size_t)TT * BB * 3 * HH * 2);
  float*  pen       = (float*) alloc((size_t)TT * BB * EE * 4);
  __bf16* pen_bf    = (__bf16*)alloc((size_t)TT * BB * EE * 2);
  float*  h0f[2], *c0f[2], *h1f[2], *c1f[2];
  __bf16* h0b[2], *h1b[2], *ctxb[2];
  for (int i = 0; i < 2; ++i) {
    h0f[i]  = (float*) alloc((size_t)BB * HH * 4);
    c0f[i]  = (float*) alloc((size_t)BB * HH * 4);
    h1f[i]  = (float*) alloc((size_t)BB * HH * 4);
    c1f[i]  = (float*) alloc((size_t)BB * HH * 4);
    h0b[i]  = (__bf16*)alloc((size_t)BB * HH * 2);
    h1b[i]  = (__bf16*)alloc((size_t)BB * HH * 2);
    ctxb[i] = (__bf16*)alloc((size_t)BB * 2 * HH * 2);
  }

  auto cvt = [&](const float* s, __bf16* d, size_t n) {
    f32_to_bf16<<<dim3((unsigned)((n + 255) / 256)), dim3(256), 0, stream>>>(s, d, n);
  };

  // ---- prologue: conversions, gather, zero-init step-0 state ----
  cvt(emb,  emb_bf,  (size_t)VV * EE);
  cvt(Wih0, Wih0_bf, (size_t)4 * HH * X0K);
  cvt(Whh0, Whh0_bf, (size_t)4 * HH * HH);
  cvt(Wih1, Wih1_bf, (size_t)4 * HH * HH);
  cvt(Whh1, Whh1_bf, (size_t)4 * HH * HH);
  cvt(Wa,   Wa_bf,   (size_t)HH * 2 * HH);
  cvt(Wp,   Wp_bf,   (size_t)EE * 3 * HH);
  cvt(enc,  enc_bf,  (size_t)SS * BB * 2 * HH);
  gather_emb<<<dim3(TT * BB), dim3(256), 0, stream>>>(trg, emb, trgemb_bf);

  hipMemsetAsync(h0b[0],  0, (size_t)BB * HH * 2,     stream);
  hipMemsetAsync(h1b[0],  0, (size_t)BB * HH * 2,     stream);
  hipMemsetAsync(c0f[0],  0, (size_t)BB * HH * 4,     stream);
  hipMemsetAsync(c1f[0],  0, (size_t)BB * HH * 4,     stream);
  hipMemsetAsync(ctxb[0], 0, (size_t)BB * 2 * HH * 2, stream);

  // ---- annot_scores = enc @ Wa^T + ba  : [S*B, H] ----
  gemm_bf16_nt<<<dim3(SS * BB / 16, HH / 64), dim3(32), 0, stream>>>(
      enc_bf, Wa_bf, ba, annot, nullptr, HH, 2 * HH);

  // ---- recurrence ----
  for (int t = 0; t < TT; ++t) {
    int p = t & 1, q = (t + 1) & 1;
    const __bf16* embt = trgemb_bf + (size_t)t * BB * EE;
    // layer 0: x = [emb_t | ctx]
    lstm_step<<<dim3(BB / 16, HH / 16), dim3(32), 0, stream>>>(
        embt, EE, EE, Wih0_bf,
        ctxb[p], 2 * HH, 2 * HH, Wih0_bf + EE, X0K,
        h0b[p], Whh0_bf, bih0, bhh0, c0f[p],
        h0f[q], h0b[q], c0f[q]);
    // layer 1: x = h0_new
    lstm_step<<<dim3(BB / 16, HH / 16), dim3(32), 0, stream>>>(
        h0b[q], HH, HH, Wih1_bf,
        nullptr, 0, 0, nullptr, HH,
        h1b[p], Whh1_bf, bih1, bhh1, c1f[p],
        h1f[q], h1b[q], c1f[q]);
    // attention + context + stacked[t]
    attn_step<<<dim3(BB), dim3(256), 0, stream>>>(
        h1f[q], Wh, bh, annot, Wo, bo, enc,
        ctxb[q],
        stacked   + (size_t)t * BB * 3 * HH,
        stackedbf + (size_t)t * BB * 3 * HH);
  }

  // ---- pen = stacked @ Wp^T + bp : [T*B, E] (also bf16 copy) ----
  gemm_bf16_nt<<<dim3(TT * BB / 16, EE / 64), dim3(32), 0, stream>>>(
      stackedbf, Wp_bf, bp, pen, pen_bf, EE, 3 * HH);

  // ---- logits = pen @ emb^T + b_out -> d_out (scores region) ----
  float* scores = (float*)d_out;
  gemm_bf16_nt<<<dim3(TT * BB / 16, VV / 64), dim3(32), 0, stream>>>(
      pen_bf, emb_bf, b_out, scores, nullptr, VV, EE);

  // ---- log_softmax in place ----
  log_softmax_rows<<<dim3(TT * BB), dim3(256), 0, stream>>>(scores, VV);

  // ---- final h, c into output tail (final state lives in buffer 0) ----
  size_t so = (size_t)TT * BB * VV;
  size_t nst = (size_t)BB * HH;
  copy_f32<<<dim3((unsigned)((nst + 255) / 256)), dim3(256), 0, stream>>>(h0f[0], scores + so,           nst);
  copy_f32<<<dim3((unsigned)((nst + 255) / 256)), dim3(256), 0, stream>>>(h1f[0], scores + so + nst,     nst);
  copy_f32<<<dim3((unsigned)((nst + 255) / 256)), dim3(256), 0, stream>>>(c0f[0], scores + so + 2 * nst, nst);
  copy_f32<<<dim3((unsigned)((nst + 255) / 256)), dim3(256), 0, stream>>>(c1f[0], scores + so + 3 * nst, nst);
}
